// MHA_86535001079889
// MI455X (gfx1250) — matrix operand order repaired and checked by reading, not compiled, not run on hardware
//
#include <hip/hip_runtime.h>
#include <hip/hip_bf16.h>

// Problem constants (match reference)
#define BB   2
#define SS   2048
#define DDIM 1024
#define HH   16
#define DKK  64

typedef __attribute__((ext_vector_type(16))) _Float16 v16h;
typedef __attribute__((ext_vector_type(8)))  _Float16 v8h;
typedef __attribute__((ext_vector_type(8)))  float    v8f;

union HF16 { v16h v; v8h p[2]; };

// lane^16 float swap via ds_swizzle_b32 (group-of-32: xor=0x10, and=0x1f)
__device__ __forceinline__ float swz_x16(float x) {
    int i = __builtin_bit_cast(int, x);
    i = __builtin_amdgcn_ds_swizzle(i, 0x401F);
    return __builtin_bit_cast(float, i);
}

// async global -> LDS copy of 16 bytes (per-lane addresses), ASYNCcnt-tracked
__device__ __forceinline__ void async_b128(unsigned ldsOff, const _Float16* gsrc) {
    asm volatile("global_load_async_to_lds_b128 %0, %1, off"
                 :: "v"(ldsOff), "v"(gsrc) : "memory");
}
__device__ __forceinline__ void wait_async0() {
    asm volatile("s_wait_asynccnt 0x0" ::: "memory");
}

// ---------------------------------------------------------------------------
// 1) fp32 -> fp16 weight conversion
// ---------------------------------------------------------------------------
__global__ void cvt_f32_f16(const float* __restrict__ src,
                            _Float16* __restrict__ dst, int n) {
    int i = blockIdx.x * blockDim.x + threadIdx.x;
    if (i < n) dst[i] = (_Float16)src[i];
}

// ---------------------------------------------------------------------------
// 2) RoPE cos/sin tables [S][DK]
// ---------------------------------------------------------------------------
__global__ void rope_tab(float* __restrict__ cosT, float* __restrict__ sinT) {
    int i = blockIdx.x * blockDim.x + threadIdx.x;   // S*32 threads
    int s = i >> 5, f = i & 31;
    float inv = expf(-((float)f / 32.0f) * logf(10000.0f));
    float ang = (float)s * inv;
    float c = cosf(ang), sn = sinf(ang);
    cosT[s * DKK + f]      = c;  cosT[s * DKK + f + 32] = c;
    sinT[s * DKK + f]      = sn; sinT[s * DKK + f + 32] = sn;
}

// ---------------------------------------------------------------------------
// 3) Projection GEMM: O = X @ W^T, f16 out in [B,H,S,DK].
//    Each wave: 32 rows x 64 cols (8 WMMAs / 32-K step). grid.z picks Q/K/V.
// ---------------------------------------------------------------------------
__global__ __launch_bounds__(256)
void proj_gemm(const float* __restrict__ Xq, const float* __restrict__ Xk,
               const float* __restrict__ Xv,
               const _Float16* __restrict__ Wq, const _Float16* __restrict__ Wk,
               const _Float16* __restrict__ Wv,
               _Float16* __restrict__ Oq, _Float16* __restrict__ Ok,
               _Float16* __restrict__ Ov) {
    const float* X; const _Float16* W; _Float16* O;
    if (blockIdx.z == 0)      { X = Xq; W = Wq; O = Oq; }
    else if (blockIdx.z == 1) { X = Xk; W = Wk; O = Ok; }
    else                      { X = Xv; W = Wv; O = Ov; }

    const int wave = threadIdx.x >> 5, lane = threadIdx.x & 31;
    const int mrow = lane & 15, hi = lane >> 4;
    const int i0 = blockIdx.x * 32;                 // M base (B*S rows)
    const int n0 = blockIdx.y * 512 + wave * 64;    // N base (D cols)

    v8f acc[2][4] = {};
    for (int kk = 0; kk < DDIM; kk += 32) {
        HF16 a[2];
#pragma unroll
        for (int mi = 0; mi < 2; ++mi) {
            const float* xrow =
                X + (size_t)(i0 + mi * 16 + mrow) * DDIM + kk + hi * 16;
#pragma unroll
            for (int e = 0; e < 8; ++e) a[mi].v[e]     = (_Float16)xrow[e];
#pragma unroll
            for (int e = 0; e < 8; ++e) a[mi].v[8 + e] = (_Float16)xrow[8 + e];
        }
#pragma unroll
        for (int nb = 0; nb < 4; ++nb) {
            const _Float16* wrow =
                W + (size_t)(n0 + nb * 16 + mrow) * DDIM + kk + hi * 16;
            v16h b = *(const v16h*)wrow;
#pragma unroll
            for (int mi = 0; mi < 2; ++mi)
                acc[mi][nb] = __builtin_amdgcn_wmma_f32_16x16x32_f16(
                    false, a[mi].v, false, b, (short)0, acc[mi][nb], false, false);
        }
    }
#pragma unroll
    for (int mi = 0; mi < 2; ++mi)
#pragma unroll
        for (int nb = 0; nb < 4; ++nb) {
            int j = n0 + nb * 16 + mrow;
            int h = j >> 6, dk = j & 63;
#pragma unroll
            for (int r = 0; r < 8; ++r) {
                int i = i0 + mi * 16 + (hi ? r + 8 : r);
                int b = i >> 11, s = i & (SS - 1);
                O[(((size_t)b * HH + h) * SS + s) * DKK + dk] =
                    (_Float16)acc[mi][nb][r];
            }
        }
}

// ---------------------------------------------------------------------------
// 4) In-place RoPE on Q or K (blockIdx.y selects buffer)
// ---------------------------------------------------------------------------
__global__ void rope_apply(_Float16* __restrict__ Q, _Float16* __restrict__ K,
                           const float* __restrict__ cosT,
                           const float* __restrict__ sinT) {
    _Float16* P = blockIdx.y ? K : Q;
    int i = blockIdx.x * blockDim.x + threadIdx.x;  // B*H*S*32 threads
    int d  = i & 31;
    int s  = (i >> 5) & (SS - 1);
    int bh = i >> 16;
    size_t base = ((size_t)bh * SS + s) * DKK;
    float x1 = (float)P[base + d], x2 = (float)P[base + d + 32];
    float c = cosT[s * DKK + d], sn = sinT[s * DKK + d];
    P[base + d]      = (_Float16)(x1 * c - x2 * sn);
    P[base + d + 32] = (_Float16)(x2 * c + x1 * sn);
}

// ---------------------------------------------------------------------------
// 5) Causal flash attention, double-buffered async K staging.
//    Grid: (S/128, B*H). 8 waves, 16 q-rows each.
// ---------------------------------------------------------------------------
__global__ __launch_bounds__(256)
void flash_attn(const _Float16* __restrict__ Qr, const _Float16* __restrict__ Kr,
                const _Float16* __restrict__ Vr, _Float16* __restrict__ AO) {
    __shared__ _Float16 Ksh[2][32][64];   // K chunk   [buf][kv][dk]  (8 KB)
    __shared__ _Float16 VTsh[2][64][32];  // V^T chunk [buf][dk][kv]  (8 KB)
    __shared__ float    Ssc[8][16][32];   // per-wave score scratch   (16 KB)
    __shared__ _Float16 Psh[8][16][32];   // per-wave P scratch       (8 KB)
    __shared__ float    rowS[8][16];      // per-wave row scales

    const int wave = threadIdx.x >> 5, lane = threadIdx.x & 31;
    const int mrow = lane & 15, hi = lane >> 4;
    const int bh = blockIdx.y;
    const int q0 = blockIdx.x * 128 + wave * 16;
    const size_t bhBase = (size_t)bh * SS * DKK;

    const int tid = threadIdx.x;
    const int ldRow = tid >> 3;            // 0..31
    const int ldCol = (tid & 7) * 8;       // 0..56

    // Q fragments, pre-scaled by 1/sqrt(DK) = 1/8
    HF16 qf[2];
#pragma unroll
    for (int dc = 0; dc < 2; ++dc) {
        const _Float16* qrow =
            Qr + bhBase + (size_t)(q0 + mrow) * DKK + dc * 32 + hi * 16;
#pragma unroll
        for (int e = 0; e < 8; ++e)
            qf[dc].v[e]     = (_Float16)((float)qrow[e] * 0.125f);
#pragma unroll
        for (int e = 0; e < 8; ++e)
            qf[dc].v[8 + e] = (_Float16)((float)qrow[8 + e] * 0.125f);
    }

    v8f o[4] = {};
    float mprev = -3.0e38f, lsum = 0.0f;   // row state (row = mrow, both halves)

    const int kvEnd = blockIdx.x * 128 + 128;

    // ---- stage chunk 0 into buffer 0 ----
    {
        const _Float16* ksrc = Kr + bhBase + (size_t)ldRow * DKK + ldCol;
        async_b128((unsigned)(size_t)&Ksh[0][ldRow][ldCol], ksrc);
        const _Float16* vsrc = Vr + bhBase + (size_t)ldRow * DKK + ldCol;
        v8h vv = *(const v8h*)vsrc;
#pragma unroll
        for (int e = 0; e < 8; ++e) VTsh[0][ldCol + e][ldRow] = vv[e];
    }
    wait_async0();
    __syncthreads();

    for (int kv = 0; kv < kvEnd; kv += 32) {
        const int cur = (kv >> 5) & 1;
        const int nxt = cur ^ 1;

        // ---- prefetch next chunk into the other buffer (async) ----
        if (kv + 32 < kvEnd) {
            const _Float16* ksrc =
                Kr + bhBase + (size_t)(kv + 32 + ldRow) * DKK + ldCol;
            async_b128((unsigned)(size_t)&Ksh[nxt][ldRow][ldCol], ksrc);
            const _Float16* vsrc =
                Vr + bhBase + (size_t)(kv + 32 + ldRow) * DKK + ldCol;
            v8h vv = *(const v8h*)vsrc;
#pragma unroll
            for (int e = 0; e < 8; ++e) VTsh[nxt][ldCol + e][ldRow] = vv[e];
        }

        if (kv <= q0 + 15) {               // wave-uniform causal skip
            // S = Q * K^T  (two 16x16 score tiles)
            v8f sacc[2] = {};
#pragma unroll
            for (int ks = 0; ks < 2; ++ks)
#pragma unroll
                for (int dc = 0; dc < 2; ++dc) {
                    v16h bfr =
                        *(const v16h*)&Ksh[cur][ks * 16 + mrow][dc * 32 + hi * 16];
                    sacc[ks] = __builtin_amdgcn_wmma_f32_16x16x32_f16(
                        false, qf[dc].v, false, bfr, (short)0, sacc[ks],
                        false, false);
                }
            // dump scores row-major into LDS
#pragma unroll
            for (int ks = 0; ks < 2; ++ks)
#pragma unroll
                for (int r = 0; r < 8; ++r)
                    Ssc[wave][hi ? r + 8 : r][ks * 16 + mrow] = sacc[ks][r];

            // online softmax: all 32 lanes. lane owns row=mrow, cols hi*16..+15
            {
                const int qg = q0 + mrow;
                const int cb = hi * 16;
                float sc[16];
                float mloc = -3.0e38f;
#pragma unroll
                for (int c = 0; c < 16; ++c) {
                    float s_ = (kv + cb + c > qg) ? -1000.0f
                                                  : Ssc[wave][mrow][cb + c];
                    sc[c] = s_;
                    mloc = fmaxf(mloc, s_);
                }
                mloc = fmaxf(mloc, swz_x16(mloc));     // combine halves
                float mnew = fmaxf(mprev, mloc);
                float fac = __expf(mprev - mnew);
                float rs = 0.0f;
#pragma unroll
                for (int c = 0; c < 16; ++c) {
                    float p = __expf(sc[c] - mnew);
                    rs += p;
                    Psh[wave][mrow][cb + c] = (_Float16)p;
                }
                rs += swz_x16(rs);                      // full row sum
                lsum = lsum * fac + rs;
                mprev = mnew;
                if (hi == 0) rowS[wave][mrow] = fac;
            }
            // rescale accumulators (same-wave LDS ops are in-order)
#pragma unroll
            for (int r = 0; r < 8; ++r) {
                float fac = rowS[wave][hi ? r + 8 : r];
#pragma unroll
                for (int nb = 0; nb < 4; ++nb) o[nb][r] *= fac;
            }
            // P fragment (16x32 A operand) from LDS
            HF16 pf;
            {
                const _Float16* prow = &Psh[wave][mrow][hi * 16];
#pragma unroll
                for (int e = 0; e < 8; ++e) pf.v[e]     = prow[e];
#pragma unroll
                for (int e = 0; e < 8; ++e) pf.v[8 + e] = prow[8 + e];
            }
            // O += P @ V
#pragma unroll
            for (int nb = 0; nb < 4; ++nb) {
                v16h vfr = *(const v16h*)&VTsh[cur][nb * 16 + mrow][hi * 16];
                o[nb] = __builtin_amdgcn_wmma_f32_16x16x32_f16(
                    false, pf.v, false, vfr, (short)0, o[nb], false, false);
            }
        }
        wait_async0();
        __syncthreads();
    }

    // finalize: divide by row sums, write f16 [B,H,S,DK]
    if (hi == 0) rowS[wave][mrow] = 1.0f / lsum;
#pragma unroll
    for (int r = 0; r < 8; ++r) {
        int row = hi ? r + 8 : r;
        float inv = rowS[wave][row];
        size_t obase = bhBase + (size_t)(q0 + row) * DKK;
#pragma unroll
        for (int nb = 0; nb < 4; ++nb)
            AO[obase + nb * 16 + mrow] = (_Float16)(o[nb][r] * inv);
    }
}

// ---------------------------------------------------------------------------
// 6) Output GEMM: out = attn @ Wo^T (f32 out). 32 rows x 64 cols per wave.
// ---------------------------------------------------------------------------
__global__ __launch_bounds__(256)
void out_gemm(const _Float16* __restrict__ AO, const _Float16* __restrict__ Wo,
              float* __restrict__ out) {
    const int wave = threadIdx.x >> 5, lane = threadIdx.x & 31;
    const int mrow = lane & 15, hi = lane >> 4;
    const int i0 = blockIdx.x * 32;
    const int n0 = blockIdx.y * 512 + wave * 64;

    v8f acc[2][4] = {};
    for (int kk = 0; kk < DDIM; kk += 32) {
        HF16 a[2];
        const int kc0 = kk + hi * 16;
        const int kc1 = kc0 + 8;
#pragma unroll
        for (int mi = 0; mi < 2; ++mi) {
            int iA = i0 + mi * 16 + mrow;
            int b = iA >> 11, s = iA & (SS - 1);
            const _Float16* a0 =
                AO + (((size_t)b * HH + (kc0 >> 6)) * SS + s) * DKK + (kc0 & 63);
            const _Float16* a1 =
                AO + (((size_t)b * HH + (kc1 >> 6)) * SS + s) * DKK + (kc1 & 63);
            a[mi].p[0] = *(const v8h*)a0;
            a[mi].p[1] = *(const v8h*)a1;
        }
#pragma unroll
        for (int nb = 0; nb < 4; ++nb) {
            const _Float16* wrow =
                Wo + (size_t)(n0 + nb * 16 + mrow) * DDIM + kk + hi * 16;
            v16h bfr = *(const v16h*)wrow;
#pragma unroll
            for (int mi = 0; mi < 2; ++mi)
                acc[mi][nb] = __builtin_amdgcn_wmma_f32_16x16x32_f16(
                    false, a[mi].v, false, bfr, (short)0, acc[mi][nb],
                    false, false);
        }
    }
#pragma unroll
    for (int mi = 0; mi < 2; ++mi)
#pragma unroll
        for (int nb = 0; nb < 4; ++nb) {
            int j = n0 + nb * 16 + mrow;
#pragma unroll
            for (int r = 0; r < 8; ++r) {
                int ii = i0 + mi * 16 + (hi ? r + 8 : r);
                out[(size_t)ii * DDIM + j] = acc[mi][nb][r];
            }
        }
}

// ---------------------------------------------------------------------------
extern "C" void kernel_launch(void* const* d_in, const int* in_sizes, int n_in,
                              void* d_out, int out_size, void* d_ws, size_t ws_size,
                              hipStream_t stream) {
    (void)in_sizes; (void)n_in; (void)out_size; (void)ws_size;
    const float* q  = (const float*)d_in[0];
    const float* k  = (const float*)d_in[1];
    const float* v  = (const float*)d_in[2];
    const float* Wq = (const float*)d_in[3];
    const float* Wk = (const float*)d_in[4];
    const float* Wv = (const float*)d_in[5];
    const float* Wo = (const float*)d_in[6];
    // d_in[7] (tril mask) unused: causality computed analytically.

    const size_t W_ELEM = (size_t)DDIM * DDIM;     // 1,048,576
    const size_t X_ELEM = (size_t)BB * SS * DDIM;  // 4,194,304

    _Float16* wq16 = (_Float16*)d_ws;
    _Float16* wk16 = wq16 + W_ELEM;
    _Float16* wv16 = wk16 + W_ELEM;
    _Float16* wo16 = wv16 + W_ELEM;
    _Float16* Qr   = wo16 + W_ELEM;   // [B,H,S,DK] f16
    _Float16* Kr   = Qr + X_ELEM;
    _Float16* Vr   = Kr + X_ELEM;
    _Float16* AO   = Vr + X_ELEM;
    float*    cosT = (float*)(AO + X_ELEM);
    float*    sinT = cosT + (size_t)SS * DKK;

    cvt_f32_f16<<<(int)(W_ELEM / 256), 256, 0, stream>>>(Wq, wq16, (int)W_ELEM);
    cvt_f32_f16<<<(int)(W_ELEM / 256), 256, 0, stream>>>(Wk, wk16, (int)W_ELEM);
    cvt_f32_f16<<<(int)(W_ELEM / 256), 256, 0, stream>>>(Wv, wv16, (int)W_ELEM);
    cvt_f32_f16<<<(int)(W_ELEM / 256), 256, 0, stream>>>(Wo, wo16, (int)W_ELEM);

    rope_tab<<<(SS * 32) / 256, 256, 0, stream>>>(cosT, sinT);

    proj_gemm<<<dim3((BB * SS) / 32, DDIM / 512, 3), 256, 0, stream>>>(
        q, k, v, wq16, wk16, wv16, Qr, Kr, Vr);

    rope_apply<<<dim3((BB * HH * SS * 32) / 256, 2), 256, 0, stream>>>(
        Qr, Kr, cosT, sinT);

    flash_attn<<<dim3(SS / 128, BB * HH), 256, 0, stream>>>(Qr, Kr, Vr, AO);

    out_gemm<<<dim3((BB * SS) / 32, DDIM / 512), 256, 0, stream>>>(
        AO, wo16, (float*)d_out);
}
